// SetConv_58677843198342
// MI455X (gfx1250) — matrix operand order repaired and checked by reading, not compiled, not run on hardware
//
#include <hip/hip_runtime.h>

// ---------------------------------------------------------------------------
// SetConv (Gaussian-RBF attention) for MI455X / gfx1250, wave32 + WMMA f16.
//   logit2[q,k] = log2e*logit = a_k + b_q + qx*gkx + qy*gky  (params pre-scaled)
//   S = exp2(logit2);  targets = (S @ V) / rowsum(S)  via v_wmma_f32_16x16x32_f16
// Split-K=4 for occupancy (4096 waves), register double-buffered B fragments,
// density rowsum via an extra WMMA against an all-ones B matrix.
// ---------------------------------------------------------------------------

typedef _Float16 v16h __attribute__((ext_vector_type(16)));
typedef float    v8f  __attribute__((ext_vector_type(8)));

__device__ __forceinline__ unsigned pack_f16x2(float a, float b) {
    auto p = __builtin_amdgcn_cvt_pkrtz(a, b);   // v_cvt_pk_rtz_f16_f32
    return __builtin_bit_cast(unsigned, p);
}

#define BATCH  8
#define NQ     2048
#define NK     2048
#define CIN    64
#define COUT   64
#define NCHUNK (NK / 32)          // 64 k-chunks of 32
#define SPLIT  4                  // k-splits per tile (16 chunks each)
#define NTILE  (BATCH * NQ / 16)  // 1024 16-query tiles

// ws layout (bytes)
#define WS_VFRAG   0u                          // 2 MB
#define WS_KPARAM  0x200000u                   // 256 KB
#define WS_PNUM    0x240000u                   // 1024*4*16*64*4 = 16 MB
#define WS_PDEN    0x1240000u                  // 1024*4*16*4   = 256 KB
#define WS_TARGETS 0x1280000u                  // 4 MB
#define WS_DCHAN   0x1680000u                  // 64 KB

__device__ __forceinline__ void rbf_params(const float* __restrict__ W,
                                           const float* __restrict__ lsp,
                                           float& c2, float& G00, float& G01, float& G11) {
    float w00 = W[0], w01 = W[1], w10 = W[2], w11 = W[3];
    G00 = w00 * w00 + w10 * w10;            // G = W^T W
    G01 = w00 * w01 + w10 * w11;
    G11 = w01 * w01 + w11 * w11;
    float x   = lsp[0] * 0.1f - 1.0f;
    float sig = 1e-5f + (x > 20.f ? x : log1pf(expf(x)));   // softplus
    c2 = -0.7213475204444817f / (sig * sig);  // -0.5*log2(e)/sigma^2
}

// --- Pre-pass 1: V (f32) -> f16 in exact WMMA B-fragment order -------------
// dword index = ((((b*64 + chunk)*4 + cblk)*32 + lane)*8 + j)
// element: k = chunk*32 + 8*(lane>=16) + (j<4 ? 2j : 16+2(j-4)) + p, c = cblk*16 + lane%16 (p=0 low)
__global__ void __launch_bounds__(256) k_vfrag(const float* __restrict__ V,
                                               unsigned* __restrict__ vfrag) {
    unsigned idx   = blockIdx.x * 256u + threadIdx.x;
    unsigned j     = idx & 7u;
    unsigned lane  = (idx >> 3) & 31u;
    unsigned cblk  = (idx >> 8) & 3u;
    unsigned chunk = (idx >> 10) & 63u;
    unsigned b     = idx >> 16;
    unsigned k     = chunk * 32u + ((lane >> 4) ? 8u : 0u) + (j < 4u ? 2u * j : 16u + 2u * (j - 4u));
    unsigned cc    = cblk * 16u + (lane & 15u);
    float f0 = V[((size_t)b * NK + k) * CIN + cc];
    float f1 = V[((size_t)b * NK + k + 1u) * CIN + cc];
    vfrag[idx] = pack_f16x2(f0, f1);
}

// --- Pre-pass 2: per-key RBF params {gx, gy, a, 0}, pre-scaled by log2e ----
__global__ void __launch_bounds__(256) k_kparams(const float* __restrict__ keys,
                                                 const float* __restrict__ W,
                                                 const float* __restrict__ lsp,
                                                 float4* __restrict__ kp) {
    int i = blockIdx.x * 256 + threadIdx.x;       // i = b*NK + k
    float c2, G00, G01, G11;
    rbf_params(W, lsp, c2, G00, G01, G11);
    float kx = keys[2 * i], ky = keys[2 * i + 1];
    float hx = G00 * kx + G01 * ky;
    float hy = G01 * kx + G11 * ky;
    float4 o;
    o.x = -2.f * c2 * hx;
    o.y = -2.f * c2 * hy;
    o.z = c2 * (kx * hx + ky * hy);
    o.w = 0.f;
    kp[i] = o;
}

// --- helpers for the main kernel -------------------------------------------
struct bfrag { uint4 lo, hi; };                  // one 32x16 f16 B operand
struct bquad { bfrag f[4]; };                    // 4 c-blocks (64 channels)

__device__ __forceinline__ v16h as_v16h(const bfrag& f) {
    return __builtin_bit_cast(v16h, f);
}
__device__ __forceinline__ void loadB(bquad& B, const unsigned* __restrict__ p) {
#pragma unroll
    for (int n = 0; n < 4; ++n) {
        B.f[n].lo = *(const uint4*)(p + n * 256);
        B.f[n].hi = *(const uint4*)(p + n * 256 + 4);
    }
}
// Build S tile in A-fragment layout: lane = row q; K runs 8h+{0..7} (VGPR 0..3)
// and 16+8h+{0..7} (VGPR 4..7); low-K half in bits [15:0].
__device__ __forceinline__ v16h computeA(const float4* __restrict__ kpb, int chunk,
                                         int h, float qx, float qy, float bq) {
    union { v16h v; unsigned d[8]; } A;
    int kbase = chunk * 32 + 8 * h;
#pragma unroll
    for (int j = 0; j < 8; ++j) {
        int kk = kbase + (j < 4 ? 2 * j : 16 + 2 * (j - 4));
        float4 k0 = kpb[kk];
        float4 k1 = kpb[kk + 1];
        float l0 = fmaf(qy, k0.y, fmaf(qx, k0.x, k0.z + bq));
        float l1 = fmaf(qy, k1.y, fmaf(qx, k1.x, k1.z + bq));
        float s0 = __builtin_amdgcn_exp2f(l0);   // logit2 <= 0: no overflow
        float s1 = __builtin_amdgcn_exp2f(l1);
        A.d[j] = pack_f16x2(s0, s1);
    }
    return A.v;
}

// --- Main: 1 wave per (16-query tile, k-split of 512 keys) -----------------
__global__ void __launch_bounds__(128) k_main(const float* __restrict__ Q,
                                              const float* __restrict__ W,
                                              const float* __restrict__ lsp,
                                              const float4* __restrict__ kp,
                                              const unsigned* __restrict__ vfrag,
                                              float* __restrict__ pnum,
                                              float* __restrict__ pden) {
    int wave  = threadIdx.x >> 5;
    int ln    = threadIdx.x & 31;
    int split = blockIdx.x >> 8;                   // 0..3
    int tile  = (blockIdx.x & 255) * 4 + wave;     // 0..1023 (block: same split)
    int b     = tile >> 7;
    int q0    = (tile & 127) << 4;
    int h     = ln >> 4;
    int qr    = ln & 15;
    int q     = q0 + qr;

    float c2, G00, G01, G11;
    rbf_params(W, lsp, c2, G00, G01, G11);
    float qx = Q[((size_t)b * NQ + q) * 2];
    float qy = Q[((size_t)b * NQ + q) * 2 + 1];
    float bq = c2 * (G00 * qx * qx + 2.f * G01 * qx * qy + G11 * qy * qy);

    v8f acc0 = {}, acc1 = {}, acc2 = {}, acc3 = {}, den = {};
    v16h ones;
#pragma unroll
    for (int i = 0; i < 16; ++i) ones[i] = (_Float16)1.0f;

    const float4*   kpb = kp + (size_t)b * NK;
    const unsigned* vfb = vfrag + (size_t)b * NCHUNK * 1024 + ln * 8;

    const int c0 = split * (NCHUNK / SPLIT);       // 16 chunks per split
    const int cE = c0 + (NCHUNK / SPLIT);

    bquad Ba, Bb;
    loadB(Ba, vfb + c0 * 1024);                    // prologue

#pragma unroll 1
    for (int ch = c0; ch < cE; ch += 2) {
        // ---- even chunk: load next B, long VALU exp chain covers latency ----
        loadB(Bb, vfb + (ch + 1) * 1024);
        __builtin_prefetch(kpb + (ch + 1) * 32, 0, 1);
        v16h A0 = computeA(kpb, ch, h, qx, qy, bq);
        acc0 = __builtin_amdgcn_wmma_f32_16x16x32_f16(false, A0, false, as_v16h(Ba.f[0]), (short)0, acc0, false, false);
        acc1 = __builtin_amdgcn_wmma_f32_16x16x32_f16(false, A0, false, as_v16h(Ba.f[1]), (short)0, acc1, false, false);
        acc2 = __builtin_amdgcn_wmma_f32_16x16x32_f16(false, A0, false, as_v16h(Ba.f[2]), (short)0, acc2, false, false);
        acc3 = __builtin_amdgcn_wmma_f32_16x16x32_f16(false, A0, false, as_v16h(Ba.f[3]), (short)0, acc3, false, false);
        den  = __builtin_amdgcn_wmma_f32_16x16x32_f16(false, A0, false, ones,            (short)0, den,  false, false);

        // ---- odd chunk: refill Ba for ch+2 (wrap-safe), consume Bb ----------
        int nc = (ch + 2 < cE) ? (ch + 2) : c0;
        loadB(Ba, vfb + nc * 1024);
        __builtin_prefetch(kpb + nc * 32, 0, 1);
        v16h A1 = computeA(kpb, ch + 1, h, qx, qy, bq);
        acc0 = __builtin_amdgcn_wmma_f32_16x16x32_f16(false, A1, false, as_v16h(Bb.f[0]), (short)0, acc0, false, false);
        acc1 = __builtin_amdgcn_wmma_f32_16x16x32_f16(false, A1, false, as_v16h(Bb.f[1]), (short)0, acc1, false, false);
        acc2 = __builtin_amdgcn_wmma_f32_16x16x32_f16(false, A1, false, as_v16h(Bb.f[2]), (short)0, acc2, false, false);
        acc3 = __builtin_amdgcn_wmma_f32_16x16x32_f16(false, A1, false, as_v16h(Bb.f[3]), (short)0, acc3, false, false);
        den  = __builtin_amdgcn_wmma_f32_16x16x32_f16(false, A1, false, ones,            (short)0, den,  false, false);
    }

    // Epilogue: store raw partials (D layout: lane = col c, rows q0 + r + 8h)
    float* pn = pnum + ((size_t)(tile * SPLIT + split) * 16) * 64;
#pragma unroll
    for (int r = 0; r < 8; ++r) {
        int ql = r + 8 * h;
        pn[(size_t)ql * 64 +  0 + qr] = acc0[r];
        pn[(size_t)ql * 64 + 16 + qr] = acc1[r];
        pn[(size_t)ql * 64 + 32 + qr] = acc2[r];
        pn[(size_t)ql * 64 + 48 + qr] = acc3[r];
    }
    if (qr == 0) {
        float* pd = pden + (size_t)(tile * SPLIT + split) * 16;
#pragma unroll
        for (int r = 0; r < 8; ++r) pd[r + 8 * h] = den[r];
    }
}

// --- Combine splits: targets = sum(num)/sum(den); density channel ----------
__global__ void __launch_bounds__(256) k_combine(const float* __restrict__ pnum,
                                                 const float* __restrict__ pden,
                                                 const float* __restrict__ dW,
                                                 const float* __restrict__ db,
                                                 float* __restrict__ targets,
                                                 float* __restrict__ dchan) {
    int idx  = blockIdx.x * 256 + threadIdx.x;   // idx = row*64 + c
    int cch  = idx & 63;
    int row  = idx >> 6;                          // row = tile*16 + ql
    int tile = row >> 4;
    int ql   = row & 15;
    const float* pn = pnum + ((size_t)tile * SPLIT * 16 + ql) * 64 + cch;
    const float* pd = pden + (size_t)tile * SPLIT * 16 + ql;
    float n = pn[0] + pn[1024] + pn[2048] + pn[3072];   // stride 16*64 per split
    float d = pd[0] + pd[16]   + pd[32]   + pd[48];
    targets[idx] = n / d;
    if (cch == 0) {
        float z = (-d * dW[0] + db[0]) * 0.1f;
        dchan[row] = 1.0f / (1.0f + __expf(-z));
    }
}

// --- Final resizer: out = [targets, dchan] @ Rw^T + rb ---------------------
__global__ void __launch_bounds__(256) k_resizer(const float* __restrict__ targets,
                                                 const float* __restrict__ dchan,
                                                 const float* __restrict__ Rw,
                                                 const float* __restrict__ rb,
                                                 float* __restrict__ out) {
    int idx = blockIdx.x * 256 + threadIdx.x;   // idx = row*COUT + o
    int o   = idx & 63;
    int row = idx >> 6;
    const float* t = targets + (size_t)row * CIN;
    const float* w = Rw + o * (CIN + 1);
    float acc = fmaf(dchan[row], w[CIN], rb[o]);
#pragma unroll 8
    for (int cc = 0; cc < CIN; ++cc) acc = fmaf(t[cc], w[cc], acc);
    out[idx] = acc;
}

extern "C" void kernel_launch(void* const* d_in, const int* in_sizes, int n_in,
                              void* d_out, int out_size, void* d_ws, size_t ws_size,
                              hipStream_t stream) {
    (void)in_sizes; (void)n_in; (void)out_size; (void)ws_size;
    const float* keys    = (const float*)d_in[0];
    const float* queries = (const float*)d_in[1];
    const float* values  = (const float*)d_in[2];
    const float* W       = (const float*)d_in[3];
    const float* lsp     = (const float*)d_in[4];
    const float* dW      = (const float*)d_in[5];
    const float* db      = (const float*)d_in[6];
    const float* Rw      = (const float*)d_in[7];
    const float* rb      = (const float*)d_in[8];

    char*     ws      = (char*)d_ws;
    unsigned* vfrag   = (unsigned*)(ws + WS_VFRAG);
    float4*   kp      = (float4*)(ws + WS_KPARAM);
    float*    pnum    = (float*)(ws + WS_PNUM);
    float*    pden    = (float*)(ws + WS_PDEN);
    float*    targets = (float*)(ws + WS_TARGETS);
    float*    dchan   = (float*)(ws + WS_DCHAN);
    float*    out     = (float*)d_out;

    k_vfrag  <<<2048, 256, 0, stream>>>(values, vfrag);            // 524288 dwords
    k_kparams<<<  64, 256, 0, stream>>>(keys, W, lsp, kp);         // 16384 keys
    k_main   <<<1024, 128, 0, stream>>>(queries, W, lsp, kp, vfrag, pnum, pden);
    k_combine<<<4096, 256, 0, stream>>>(pnum, pden, dW, db, targets, dchan);
    k_resizer<<<4096, 256, 0, stream>>>(targets, dchan, Rw, rb, out);
}
